// MemEffAttn_67259187855883
// MI455X (gfx1250) — matrix operand order repaired and checked by reading, not compiled, not run on hardware
//
#include <hip/hip_runtime.h>
#include <hip/hip_bf16.h>

typedef __attribute__((ext_vector_type(16))) _Float16 v16h;
typedef __attribute__((ext_vector_type(8)))  _Float16 v8h;
typedef __attribute__((ext_vector_type(8)))  float    v8f;
typedef __attribute__((ext_vector_type(2)))  float    v2f;

#define TDIM 256
#define BATCH 8
#define EDIM 512
#define NH 32
#define HD 16
#define EXPN 512
#define SLEN 768   // T + EXP
#define MROWS 2048 // T*B
#define SMOOTH_C 20.0f

// ---------------- fp32 -> fp16 convert ----------------
__global__ __launch_bounds__(256) void cvt_f32_f16(const float* __restrict__ src,
                                                   _Float16* __restrict__ dst, int n) {
    int i = blockIdx.x * 256 + threadIdx.x;
    if (i < n) dst[i] = (_Float16)src[i];
}

// ---------------- f16 WMMA GEMM:  C[m][n] = sum_k A[m][k] * W[n][k] (+bias[n]) ----------------
// A: M x K f16 row-major. W: N x K f16 row-major (original jax weight layout [f][e]).
// Block: 128 threads = 4 waves. Each wave owns a 16x64 tile (A-fragment reused across
// 4 B-fragments -> fewer loads per WMMA). Block covers 64 rows x 64 cols.
__global__ __launch_bounds__(128) void gemm16_kernel(const _Float16* __restrict__ A,
                                                     const _Float16* __restrict__ W,
                                                     const float* __restrict__ bias,
                                                     float* __restrict__ C,
                                                     int M, int N, int K) {
    const int tid  = threadIdx.x;
    const int wave = tid >> 5;
    const int lane = tid & 31;
    const int half = lane >> 4;      // 0: lanes 0-15, 1: lanes 16-31
    const int l15  = lane & 15;
    const int m0 = blockIdx.x * 64 + wave * 16;
    const int n0 = blockIdx.y * 64;

    v8f c[4] = {{}, {}, {}, {}};
    const _Float16* arow  = A + (size_t)(m0 + l15) * K;  // A-row for this lane
    const _Float16* wrow0 = W + (size_t)(n0 + l15) * K;  // B-col for this lane (W row)

    for (int kk = 0; kk < K; kk += 32) {
        // A 16x32 f16 layout: lanes<16 row=l15 K={0..7,16..23}; lanes>=16 K={8..15,24..31}
        v8h alo = *(const v8h*)(arow + kk + half * 8);
        v8h ahi = *(const v8h*)(arow + kk + 16 + half * 8);
        v16h a;
        for (int i = 0; i < 8; i++) { a[i] = alo[i]; a[i + 8] = ahi[i]; }

        for (int nt = 0; nt < 4; nt++) {
            // B 32x16 f16 layout: lanes<16 col=l15 K={0..7,16..23}; lanes>=16 K={8..15,24..31}
            const _Float16* wr = wrow0 + (size_t)nt * 16 * K;
            v8h blo = *(const v8h*)(wr + kk + half * 8);
            v8h bhi = *(const v8h*)(wr + kk + 16 + half * 8);
            v16h bm;
            for (int i = 0; i < 8; i++) { bm[i] = blo[i]; bm[i + 8] = bhi[i]; }
            c[nt] = __builtin_amdgcn_wmma_f32_16x16x32_f16(false, a, false, bm,
                                                           (short)0, c[nt], false, false);
        }
    }
    // C/D layout: VGPR j -> row j + 8*half, col l15
    for (int nt = 0; nt < 4; nt++) {
        const float bb = bias ? bias[n0 + nt * 16 + l15] : 0.0f;
        for (int j = 0; j < 8; j++)
            C[(size_t)(m0 + j + half * 8) * N + n0 + nt * 16 + l15] = c[nt][j] + bb;
    }
}

// ---------------- gather/expand K,V into [b][s][e] ----------------
__global__ __launch_bounds__(256) void expand_kernel(const float* __restrict__ kf,
                                                     const float* __restrict__ vf,
                                                     const int* __restrict__ oc,
                                                     float* __restrict__ Kext,
                                                     float* __restrict__ Vext) {
    int idx = blockIdx.x * 256 + threadIdx.x;        // over B*SLEN*(EDIM/4)
    const int total = BATCH * SLEN * (EDIM / 4);
    if (idx >= total) return;
    int c4 = idx & 127;
    int s  = (idx >> 7) % SLEN;
    int b  = idx / (128 * SLEN);
    int src_t = (s < TDIM) ? s : oc[b * EXPN + (s - TDIM)];
    const float4* ks = (const float4*)(kf + ((size_t)src_t * BATCH + b) * EDIM) + c4;
    const float4* vs = (const float4*)(vf + ((size_t)src_t * BATCH + b) * EDIM) + c4;
    float4* kd = (float4*)(Kext + ((size_t)b * SLEN + s) * EDIM) + c4;
    float4* vd = (float4*)(Vext + ((size_t)b * SLEN + s) * EDIM) + c4;
    *kd = *ks;
    *vd = *vs;
}

__global__ __launch_bounds__(256) void mask_kernel(const unsigned char* __restrict__ kpm,
                                                   const unsigned char* __restrict__ em,
                                                   unsigned char* __restrict__ mb) {
    int i = blockIdx.x * 256 + threadIdx.x;          // over B*SLEN
    if (i >= BATCH * SLEN) return;
    int b = i / SLEN, s = i % SLEN;
    mb[i] = (s < TDIM) ? kpm[b * TDIM + s] : em[b * EXPN + (s - TDIM)];
}

// ---------------- flash attention per (b, h, t-block) with f32 WMMA ----------------
// grid: (T/64, H, B); block 128 (4 waves, each wave one 16-row t tile).
// K/V head-slices staged into LDS with gfx1250 async-copy (ASYNCcnt-tracked DMA).
__global__ __launch_bounds__(128) void attn_kernel(const float* __restrict__ qf,
                                                   const float* __restrict__ Kext,
                                                   const float* __restrict__ Vext,
                                                   const float* __restrict__ law,
                                                   const unsigned char* __restrict__ mb,
                                                   float* __restrict__ attnOut) {
    __shared__ float Ks[64 * HD];
    __shared__ float Vs[64 * HD];
    __shared__ float Pb[4 * 256];

    const int tid  = threadIdx.x;
    const int wave = tid >> 5;
    const int lane = tid & 31;
    const int half = lane >> 4;
    const int l15  = lane & 15;
    const int h = blockIdx.y;
    const int b = blockIdx.z;
    const int t0 = blockIdx.x * 64 + wave * 16;

    // low 32 bits of a flat shared-aperture address == LDS byte offset
    const unsigned ksBase = (unsigned)(size_t)(void*)Ks;
    const unsigned vsBase = (unsigned)(size_t)(void*)Vs;

    // Q A-frags (f32 16x4): lanes<16 row=l15 K={4kk,4kk+1}; lanes>=16 K={4kk+2,4kk+3}
    v2f qa[4];
    {
        const float* qrow = qf + ((size_t)(t0 + l15) * BATCH + b) * EDIM + h * HD;
        for (int kk = 0; kk < 4; kk++) {
            int d = kk * 4 + half * 2;
            qa[kk].x = qrow[d];
            qa[kk].y = qrow[d + 1];
        }
    }

    float mrow[8], lrow[8];
    v8f acc = {};
    for (int j = 0; j < 8; j++) { mrow[j] = -__builtin_inff(); lrow[j] = 0.0f; }

    const float* Kbh = Kext + (size_t)b * SLEN * EDIM + h * HD;
    const float* Vbh = Vext + (size_t)b * SLEN * EDIM + h * HD;

    for (int scn = 0; scn < SLEN / 64; scn++) {
        const int sbase = scn * 64;
        __syncthreads();   // previous P*V reads of Vs are done before overwrite
        // stage K,V chunk (64 x 16 f32 each) into LDS via async copy; 128 threads, 2 x 16B each
        for (int i = 0; i < 2; i++) {
            int idx = tid * 2 + i;                  // 0..255
            int r = idx >> 2, c = idx & 3;
            const float* ksrc = Kbh + (size_t)(sbase + r) * EDIM + c * 4;
            const float* vsrc = Vbh + (size_t)(sbase + r) * EDIM + c * 4;
            unsigned kdst = ksBase + (unsigned)idx * 16u;
            unsigned vdst = vsBase + (unsigned)idx * 16u;
            asm volatile("global_load_async_to_lds_b128 %0, %1, off"
                         :: "v"(kdst), "v"(ksrc) : "memory");
            asm volatile("global_load_async_to_lds_b128 %0, %1, off"
                         :: "v"(vdst), "v"(vsrc) : "memory");
            if (scn + 1 < SLEN / 64)                // hint next chunk toward the caches
                __builtin_prefetch(Kbh + (size_t)(sbase + 64 + r) * EDIM + c * 4, 0, 1);
        }
        asm volatile("s_wait_asynccnt 0" ::: "memory");
        __syncthreads();

        for (int st = 0; st < 4; st++) {
            const int sl = st * 16;
            const int s0 = sbase + sl;

            // scores tile (16 t-rows x 16 s-cols) = Q(16x16) * K^T via 4x WMMA f32 16x16x4
            v8f c8 = {};
            for (int kk = 0; kk < 4; kk++) {
                v2f kb;
                int col = sl + l15;
                int d = kk * 4 + half * 2;
                kb.x = Ks[col * HD + d];
                kb.y = Ks[col * HD + d + 1];
                c8 = __builtin_amdgcn_wmma_f32_16x16x4_f32(false, qa[kk], false, kb,
                                                           (short)0, c8, false, false);
            }

            // smoothing * law, masking; online softmax update
            const int col = s0 + l15;
            const unsigned char mk = mb[b * SLEN + col];
            float pv[8];
            for (int j = 0; j < 8; j++) {
                const int trow = t0 + j + half * 8;
                const float lv = law[((size_t)(b * TDIM + trow)) * SLEN + col];
                float s = (c8[j] + SMOOTH_C) * lv - SMOOTH_C;
                if (mk) s = -__builtin_inff();

                float r = s;
                r = fmaxf(r, __shfl_xor(r, 1));
                r = fmaxf(r, __shfl_xor(r, 2));
                r = fmaxf(r, __shfl_xor(r, 4));
                r = fmaxf(r, __shfl_xor(r, 8));
                const float mnew  = fmaxf(mrow[j], r);
                const float msafe = (mnew == -__builtin_inff()) ? 0.0f : mnew;
                const float scale = (mrow[j] == -__builtin_inff()) ? 0.0f
                                                                   : __expf(mrow[j] - msafe);
                const float e = (s == -__builtin_inff()) ? 0.0f : __expf(s - msafe);
                float es = e;
                es += __shfl_xor(es, 1);
                es += __shfl_xor(es, 2);
                es += __shfl_xor(es, 4);
                es += __shfl_xor(es, 8);
                lrow[j] = lrow[j] * scale + es;
                mrow[j] = mnew;
                acc[j] *= scale;
                pv[j] = e * lv;           // probs *= law (post-softmax numerator)
            }

            // re-stripe P (C-layout -> A-layout) through LDS
            float* Pw = Pb + wave * 256;
            for (int j = 0; j < 8; j++)
                Pw[(j + half * 8) * 16 + l15] = pv[j];
            __syncthreads();

            // acc += P(16x16) * V(16x16) via 4x WMMA f32 16x16x4
            for (int kk = 0; kk < 4; kk++) {
                v2f pa, vb;
                const int d2 = kk * 4 + half * 2;
                pa.x = Pw[l15 * 16 + d2];
                pa.y = Pw[l15 * 16 + d2 + 1];
                vb.x = Vs[(sl + d2) * HD + l15];
                vb.y = Vs[(sl + d2 + 1) * HD + l15];
                acc = __builtin_amdgcn_wmma_f32_16x16x4_f32(false, pa, false, vb,
                                                            (short)0, acc, false, false);
            }
            __syncthreads();
        }
    }

    for (int j = 0; j < 8; j++) {
        const int trow = t0 + j + half * 8;
        const float denom = lrow[j];
        const float o = (denom > 0.0f) ? acc[j] / denom : 0.0f;
        attnOut[((size_t)trow * BATCH + b) * EDIM + h * HD + l15] = o;
    }
}

// ---------------- host-side launch ----------------
extern "C" void kernel_launch(void* const* d_in, const int* in_sizes, int n_in,
                              void* d_out, int out_size, void* d_ws, size_t ws_size,
                              hipStream_t stream) {
    const float*         query = (const float*)d_in[0];
    const int*           ocidx = (const int*)d_in[1];
    const unsigned char* emask = (const unsigned char*)d_in[2];
    const unsigned char* kpm   = (const unsigned char*)d_in[3];
    const float*         law   = (const float*)d_in[4];
    const float*         Wq    = (const float*)d_in[5];
    const float*         bq    = (const float*)d_in[6];
    const float*         Wk    = (const float*)d_in[7];
    const float*         Wv    = (const float*)d_in[8];
    const float*         bv    = (const float*)d_in[9];
    const float*         Wo    = (const float*)d_in[10];
    const float*         bo    = (const float*)d_in[11];
    float* out = (float*)d_out;

    char* ws = (char*)d_ws;
    size_t off = 0;
    auto take = [&](size_t bytes) { char* p = ws + off; off += (bytes + 255) & ~(size_t)255; return p; };

    _Float16* A16   = (_Float16*)take((size_t)MROWS * EDIM * 2);
    _Float16* W16q  = (_Float16*)take((size_t)EDIM * EDIM * 2);
    _Float16* W16k  = (_Float16*)take((size_t)EDIM * EDIM * 2);
    _Float16* W16v  = (_Float16*)take((size_t)EDIM * EDIM * 2);
    _Float16* W16o  = (_Float16*)take((size_t)EDIM * EDIM * 2);
    float*    qf    = (float*)take((size_t)MROWS * EDIM * 4);
    float*    kf    = (float*)take((size_t)MROWS * EDIM * 4);
    float*    vf    = (float*)take((size_t)MROWS * EDIM * 4);
    float*    Kext  = (float*)take((size_t)BATCH * SLEN * EDIM * 4);
    float*    Vext  = (float*)take((size_t)BATCH * SLEN * EDIM * 4);
    float*    attnF = (float*)take((size_t)MROWS * EDIM * 4);
    _Float16* attn16= (_Float16*)take((size_t)MROWS * EDIM * 2);
    unsigned char* mbuf = (unsigned char*)take((size_t)BATCH * SLEN);

    const int nQ = MROWS * EDIM;     // 1,048,576
    const int nW = EDIM * EDIM;      // 262,144

    cvt_f32_f16<<<(nQ + 255) / 256, 256, 0, stream>>>(query, A16, nQ);
    cvt_f32_f16<<<(nW + 255) / 256, 256, 0, stream>>>(Wq, W16q, nW);
    cvt_f32_f16<<<(nW + 255) / 256, 256, 0, stream>>>(Wk, W16k, nW);
    cvt_f32_f16<<<(nW + 255) / 256, 256, 0, stream>>>(Wv, W16v, nW);
    cvt_f32_f16<<<(nW + 255) / 256, 256, 0, stream>>>(Wo, W16o, nW);

    dim3 ggrid(MROWS / 64, EDIM / 64);   // (32, 8)
    gemm16_kernel<<<ggrid, 128, 0, stream>>>(A16, W16q, bq,      qf, MROWS, EDIM, EDIM);
    gemm16_kernel<<<ggrid, 128, 0, stream>>>(A16, W16k, nullptr, kf, MROWS, EDIM, EDIM);
    gemm16_kernel<<<ggrid, 128, 0, stream>>>(A16, W16v, bv,      vf, MROWS, EDIM, EDIM);

    const int nExp = BATCH * SLEN * (EDIM / 4);
    expand_kernel<<<(nExp + 255) / 256, 256, 0, stream>>>(kf, vf, ocidx, Kext, Vext);
    mask_kernel<<<(BATCH * SLEN + 255) / 256, 256, 0, stream>>>(kpm, emask, mbuf);

    dim3 agrid(TDIM / 64, NH, BATCH);    // (4, 32, 8)
    attn_kernel<<<agrid, 128, 0, stream>>>(qf, Kext, Vext, law, mbuf, attnF);

    cvt_f32_f16<<<(nQ + 255) / 256, 256, 0, stream>>>(attnF, attn16, nQ);
    gemm16_kernel<<<ggrid, 128, 0, stream>>>(attn16, W16o, bo, out, MROWS, EDIM, EDIM);
}
